// BaseODERNN_24678882083394
// MI455X (gfx1250) — matrix operand order repaired and checked by reading, not compiled, not run on hardware
//
#include <hip/hip_runtime.h>
#include <hip/hip_bf16.h>

// ODE-RNN on MI455X (gfx1250, wave32).
// Each wave owns 16 batch rows; h state stays in VGPRs (D-layout v8f tiles)
// for the entire 199-step serial scan. All GEMMs use v_wmma_f32_16x16x32_f16.
// Weights cached in LDS (f16); W1/W2 B-fragments additionally cached in VGPRs.
// Activations use v_exp_f32 + v_rcp_f32 (no IEEE division on the critical path).

typedef __attribute__((ext_vector_type(16))) _Float16 v16h;
typedef __attribute__((ext_vector_type(8)))  float    v8f;

#define T_STEPS 200
#define BATCH   2048
#define DIN     64
#define HDIM    128
#define NCLS    32
#define MLPP    64          // MLP hidden 50 padded to 64
#define MLPH    50
#define ROWS_PER_WAVE   16
#define WAVES_PER_BLOCK 4
#define ROWS_PER_BLOCK  (ROWS_PER_WAVE * WAVES_PER_BLOCK)

union FragH { v16h v; unsigned u[8]; _Float16 e[16]; };

// tanh(x) = sign(x) * (1-e)/(1+e), e = exp(-2|x|).  Division replaced by
// v_rcp_f32 (~1 ulp; inputs are f16-quantized anyway).
__device__ __forceinline__ float ftanh(float x) {
  float e = __expf(-2.0f * fabsf(x));
  float t = (1.0f - e) * __builtin_amdgcn_rcpf(1.0f + e);
  return x >= 0.0f ? t : -t;
}
__device__ __forceinline__ float fsigmoid(float x) {
  return __builtin_amdgcn_rcpf(1.0f + __expf(-x));
}

__device__ __forceinline__ v8f wmma_f16(v16h a, v16h b, v8f c) {
  return __builtin_amdgcn_wmma_f32_16x16x32_f16(false, a, false, b, (short)0, c,
                                                false, false);
}

// A-matrix fragment (16x32 f16) from row-major f16 buffer, tile start = `tile`.
// Lane L: row = L%16; VGPR j<4: K = (L<16?0:8)+2j ; VGPR j>=4: K = 16+(L<16?0:8)+2(j-4).
__device__ __forceinline__ v16h lds_load_A(const _Float16* tile, int ld, int lane) {
  int r = lane & 15, hi = lane >> 4;
  const _Float16* p = tile + r * ld + hi * 8;
  FragH f;
#pragma unroll
  for (int j = 0; j < 4; ++j) f.u[j]     = *(const unsigned*)(p + 2 * j);
#pragma unroll
  for (int j = 0; j < 4; ++j) f.u[4 + j] = *(const unsigned*)(p + 16 + 2 * j);
  return f.v;
}

// B-matrix fragment (32x16 f16). B[k][n] = W[n][k] with W stored row-major
// [out][in]; `w` points at W[n0][k0]. Lane L: col n = L%16; VGPR j<4: K = (L<16?0:8)+2j ; VGPR j>=4: K = 16+(L<16?0:8)+2(j-4).
__device__ __forceinline__ v16h lds_load_B(const _Float16* w, int ld, int lane) {
  int n = lane & 15, hi = lane >> 4;
  const _Float16* p = w + n * ld + hi * 8;
  FragH f;
#pragma unroll
  for (int j = 0; j < 4; ++j) f.u[j]     = *(const unsigned*)(p + 2 * j);
#pragma unroll
  for (int j = 0; j < 4; ++j) f.u[4 + j] = *(const unsigned*)(p + 16 + 2 * j);
  return f.v;
}

// A fragment straight from global f32 (for x_t), converting to f16.
__device__ __forceinline__ v16h gbl_load_A_f32(const float* tile, int ld, int lane) {
  int r = lane & 15, hi = lane >> 4;
  const float* p = tile + r * ld + hi * 8;
  FragH f;
#pragma unroll
  for (int j = 0; j < 4; ++j) {
    f.e[2 * j]         = (_Float16)p[2 * j];
    f.e[2 * j + 1]     = (_Float16)p[2 * j + 1];
    f.e[8 + 2 * j]     = (_Float16)p[16 + 2 * j];
    f.e[8 + 2 * j + 1] = (_Float16)p[16 + 2 * j + 1];
  }
  return f.v;
}

// Store a C/D tile (v8f) to row-major f16: lane L holds col n=L%16,
// rows j + (L<16?0:8).
__device__ __forceinline__ void store_tile_f16(_Float16* dst, int ld, int lane, v8f d) {
  int n = lane & 15, hi = lane >> 4;
#pragma unroll
  for (int j = 0; j < 8; ++j) dst[(hi * 8 + j) * ld + n] = (_Float16)d[j];
}

__device__ __forceinline__ void write_h_tiles(_Float16* hsb, int lane, const v8f h[8]) {
  __builtin_amdgcn_wave_barrier();
#pragma unroll
  for (int j = 0; j < 8; ++j) store_tile_f16(hsb + j * 16, HDIM, lane, h[j]);
  __builtin_amdgcn_wave_barrier();
}

// k = f(hs) = tanh(hs @ W1^T + b1) @ W2^T + b2 ; hs read (f16) from hsb.
__device__ __forceinline__ void f_eval(int lane, const _Float16* hsb, _Float16* zbuf,
                                       const v16h (&w1f)[4][4], const v16h (&w2f)[8][2],
                                       const float* b1s, const float* b2s, v8f k[8]) {
  int l16 = lane & 15;
  v16h hA[4];
#pragma unroll
  for (int kt = 0; kt < 4; ++kt) hA[kt] = lds_load_A(hsb + kt * 32, HDIM, lane);
#pragma unroll
  for (int nt = 0; nt < 4; ++nt) {
    v8f acc = {};
#pragma unroll
    for (int kt = 0; kt < 4; ++kt) acc = wmma_f16(hA[kt], w1f[nt][kt], acc);
    float bb = b1s[nt * 16 + l16];
    v8f z;
#pragma unroll
    for (int c = 0; c < 8; ++c) z[c] = ftanh(acc[c] + bb);
    store_tile_f16(zbuf + nt * 16, MLPP, lane, z);
  }
  __builtin_amdgcn_wave_barrier();
  v16h zA[2];
#pragma unroll
  for (int kt = 0; kt < 2; ++kt) zA[kt] = lds_load_A(zbuf + kt * 32, MLPP, lane);
#pragma unroll
  for (int nt = 0; nt < 8; ++nt) {
    v8f acc = {};
#pragma unroll
    for (int kt = 0; kt < 2; ++kt) acc = wmma_f16(zA[kt], w2f[nt][kt], acc);
    float bb = b2s[nt * 16 + l16];
#pragma unroll
    for (int c = 0; c < 8; ++c) acc[c] += bb;
    k[nt] = acc;
  }
}

__global__ void __launch_bounds__(128, 1)
ode_rnn_kernel(const float* __restrict__ x, const float* __restrict__ tarr,
               const float* __restrict__ w1, const float* __restrict__ b1,
               const float* __restrict__ w2, const float* __restrict__ b2,
               const float* __restrict__ wih, const float* __restrict__ whh,
               const float* __restrict__ bih, const float* __restrict__ bhh,
               const float* __restrict__ fcw, const float* __restrict__ fcb,
               float* __restrict__ out) {
  extern __shared__ char smem[];
  _Float16* w1s   = (_Float16*)smem;                       // [64][128] padded
  _Float16* w2s   = w1s  + MLPP * HDIM;                    // [128][64] padded
  _Float16* wihs  = w2s  + HDIM * MLPP;                    // [384][64]
  _Float16* whhs  = wihs + 3 * HDIM * DIN;                 // [384][128]
  _Float16* fcws  = whhs + 3 * HDIM * HDIM;                // [32][128]
  _Float16* hsball = fcws + NCLS * HDIM;                   // per-wave [16][128]
  _Float16* zball  = hsball + WAVES_PER_BLOCK * ROWS_PER_WAVE * HDIM; // per-wave [16][64]
  float* b1s  = (float*)(zball + WAVES_PER_BLOCK * ROWS_PER_WAVE * MLPP);
  float* b2s  = b1s + MLPP;
  float* bihs = b2s + HDIM;
  float* bhhs = bihs + 3 * HDIM;
  float* fcbs = bhhs + 3 * HDIM;

  const int tid = threadIdx.x;
  // Cooperative weight load (f32 -> f16), zero-padding MLP dim 50 -> 64.
  for (int i = tid; i < MLPP * HDIM; i += 128) {
    int r = i / HDIM, c = i % HDIM;
    w1s[i] = (_Float16)((r < MLPH) ? w1[r * HDIM + c] : 0.0f);
  }
  for (int i = tid; i < HDIM * MLPP; i += 128) {
    int r = i / MLPP, c = i % MLPP;
    w2s[i] = (_Float16)((c < MLPH) ? w2[r * MLPH + c] : 0.0f);
  }
  for (int i = tid; i < 3 * HDIM * DIN; i += 128)  wihs[i] = (_Float16)wih[i];
  for (int i = tid; i < 3 * HDIM * HDIM; i += 128) whhs[i] = (_Float16)whh[i];
  for (int i = tid; i < NCLS * HDIM; i += 128)     fcws[i] = (_Float16)fcw[i];
  for (int i = tid; i < MLPP; i += 128)            b1s[i] = (i < MLPH) ? b1[i] : 0.0f;
  for (int i = tid; i < HDIM; i += 128)            b2s[i] = b2[i];
  for (int i = tid; i < 3 * HDIM; i += 128) { bihs[i] = bih[i]; bhhs[i] = bhh[i]; }
  for (int i = tid; i < NCLS; i += 128)            fcbs[i] = fcb[i];
  __syncthreads();

  const int wave = tid >> 5;
  const int lane = tid & 31;
  const int l16  = lane & 15;
  const int hi   = lane >> 4;
  const int row0 = blockIdx.x * ROWS_PER_BLOCK + wave * ROWS_PER_WAVE;
  _Float16* hsb  = hsball + wave * (ROWS_PER_WAVE * HDIM);
  _Float16* zbuf = zball  + wave * (ROWS_PER_WAVE * MLPP);

  // Register-resident B fragments of W1 (4 ntile x 4 ktile) and W2 (8 x 2):
  // reused 16x per step x 199 steps.
  v16h w1f[4][4], w2f[8][2];
#pragma unroll
  for (int nt = 0; nt < 4; ++nt)
#pragma unroll
    for (int kt = 0; kt < 4; ++kt)
      w1f[nt][kt] = lds_load_B(w1s + (nt * 16) * HDIM + kt * 32, HDIM, lane);
#pragma unroll
  for (int nt = 0; nt < 8; ++nt)
#pragma unroll
    for (int kt = 0; kt < 2; ++kt)
      w2f[nt][kt] = lds_load_B(w2s + (nt * 16) * MLPP + kt * 32, MLPP, lane);

  v8f h[8] = {};   // h[16,128] in D-layout tiles, f32
  v8f k[8];

#pragma unroll 1
  for (int t = 0; t < T_STEPS - 1; ++t) {
    // Prefetch next step's x rows (one 128B line per lane covers 16x64 f32).
    if (t + 1 < T_STEPS - 1) {
      const char* nx = (const char*)(x + (size_t)(t + 1) * BATCH * DIN +
                                     (size_t)row0 * DIN) + (size_t)lane * 128;
      __builtin_prefetch(nx, 0, 1);
    }

    float dt  = tarr[t + 1] - tarr[t];
    float sub = dt * 0.25f;   // N_SUB = 4

#pragma unroll 1
    for (int s = 0; s < 4; ++s) {   // RK4 substeps
      write_h_tiles(hsb, lane, h);
      f_eval(lane, hsb, zbuf, w1f, w2f, b1s, b2s, k);   // k1
      v8f hacc[8];
      __builtin_amdgcn_wave_barrier();
#pragma unroll
      for (int j = 0; j < 8; ++j) {
        hacc[j] = h[j] + (sub * (1.0f / 6.0f)) * k[j];
        v8f st = h[j] + (0.5f * sub) * k[j];
        store_tile_f16(hsb + j * 16, HDIM, lane, st);
      }
      __builtin_amdgcn_wave_barrier();
      f_eval(lane, hsb, zbuf, w1f, w2f, b1s, b2s, k);   // k2
      __builtin_amdgcn_wave_barrier();
#pragma unroll
      for (int j = 0; j < 8; ++j) {
        hacc[j] += (sub * (1.0f / 3.0f)) * k[j];
        v8f st = h[j] + (0.5f * sub) * k[j];
        store_tile_f16(hsb + j * 16, HDIM, lane, st);
      }
      __builtin_amdgcn_wave_barrier();
      f_eval(lane, hsb, zbuf, w1f, w2f, b1s, b2s, k);   // k3
      __builtin_amdgcn_wave_barrier();
#pragma unroll
      for (int j = 0; j < 8; ++j) {
        hacc[j] += (sub * (1.0f / 3.0f)) * k[j];
        v8f st = h[j] + sub * k[j];
        store_tile_f16(hsb + j * 16, HDIM, lane, st);
      }
      __builtin_amdgcn_wave_barrier();
      f_eval(lane, hsb, zbuf, w1f, w2f, b1s, b2s, k);   // k4
#pragma unroll
      for (int j = 0; j < 8; ++j) h[j] = hacc[j] + (sub * (1.0f / 6.0f)) * k[j];
    }

    // ---- GRU cell: gate order (r, z, n); n = tanh(gi_n + r * gh_n) ----
    write_h_tiles(hsb, lane, h);
    v16h hA[4];
#pragma unroll
    for (int kt = 0; kt < 4; ++kt) hA[kt] = lds_load_A(hsb + kt * 32, HDIM, lane);
    const float* xrow = x + (size_t)t * BATCH * DIN + (size_t)row0 * DIN;
    v16h xA[2];
#pragma unroll
    for (int kt = 0; kt < 2; ++kt) xA[kt] = gbl_load_A_f32(xrow + kt * 32, DIN, lane);

#pragma unroll
    for (int j = 0; j < 8; ++j) {           // column tiles of H
      int nb = j * 16 + l16;
      v8f gr = {};
#pragma unroll
      for (int kt = 0; kt < 2; ++kt)
        gr = wmma_f16(xA[kt], lds_load_B(wihs + (0 * HDIM + j * 16) * DIN + kt * 32, DIN, lane), gr);
#pragma unroll
      for (int kt = 0; kt < 4; ++kt)
        gr = wmma_f16(hA[kt], lds_load_B(whhs + (0 * HDIM + j * 16) * HDIM + kt * 32, HDIM, lane), gr);
      float brr = bihs[nb] + bhhs[nb];
      v8f r;
#pragma unroll
      for (int c = 0; c < 8; ++c) r[c] = fsigmoid(gr[c] + brr);

      v8f gz = {};
#pragma unroll
      for (int kt = 0; kt < 2; ++kt)
        gz = wmma_f16(xA[kt], lds_load_B(wihs + (1 * HDIM + j * 16) * DIN + kt * 32, DIN, lane), gz);
#pragma unroll
      for (int kt = 0; kt < 4; ++kt)
        gz = wmma_f16(hA[kt], lds_load_B(whhs + (1 * HDIM + j * 16) * HDIM + kt * 32, HDIM, lane), gz);
      float bzz = bihs[HDIM + nb] + bhhs[HDIM + nb];
      v8f zg;
#pragma unroll
      for (int c = 0; c < 8; ++c) zg[c] = fsigmoid(gz[c] + bzz);

      v8f gin = {};
#pragma unroll
      for (int kt = 0; kt < 2; ++kt)
        gin = wmma_f16(xA[kt], lds_load_B(wihs + (2 * HDIM + j * 16) * DIN + kt * 32, DIN, lane), gin);
      v8f ghn = {};
#pragma unroll
      for (int kt = 0; kt < 4; ++kt)
        ghn = wmma_f16(hA[kt], lds_load_B(whhs + (2 * HDIM + j * 16) * HDIM + kt * 32, HDIM, lane), ghn);
      float bin = bihs[2 * HDIM + nb], bhn = bhhs[2 * HDIM + nb];
      v8f hnew;
#pragma unroll
      for (int c = 0; c < 8; ++c) {
        float n = ftanh(gin[c] + bin + r[c] * (ghn[c] + bhn));
        hnew[c] = (1.0f - zg[c]) * n + zg[c] * h[j][c];
      }
      h[j] = hnew;
    }

    // ---- FC head: out[t] = h @ fc_w^T + fc_b  -> [16, 32] ----
    write_h_tiles(hsb, lane, h);
    v16h hA2[4];
#pragma unroll
    for (int kt = 0; kt < 4; ++kt) hA2[kt] = lds_load_A(hsb + kt * 32, HDIM, lane);
    float* orow = out + (size_t)t * BATCH * NCLS + (size_t)row0 * NCLS;
#pragma unroll
    for (int j2 = 0; j2 < 2; ++j2) {
      v8f oacc = {};
#pragma unroll
      for (int kt = 0; kt < 4; ++kt)
        oacc = wmma_f16(hA2[kt], lds_load_B(fcws + (j2 * 16) * HDIM + kt * 32, HDIM, lane), oacc);
      float bb = fcbs[j2 * 16 + l16];
#pragma unroll
      for (int c = 0; c < 8; ++c)
        orow[(size_t)(hi * 8 + c) * NCLS + j2 * 16 + l16] = oacc[c] + bb;
    }
  }
}

extern "C" void kernel_launch(void* const* d_in, const int* in_sizes, int n_in,
                              void* d_out, int out_size, void* d_ws, size_t ws_size,
                              hipStream_t stream) {
  (void)in_sizes; (void)n_in; (void)out_size; (void)d_ws; (void)ws_size;
  const float* x    = (const float*)d_in[0];
  const float* tarr = (const float*)d_in[1];
  const float* w1   = (const float*)d_in[2];
  const float* b1   = (const float*)d_in[3];
  const float* w2   = (const float*)d_in[4];
  const float* b2   = (const float*)d_in[5];
  const float* wih  = (const float*)d_in[6];
  const float* whh  = (const float*)d_in[7];
  const float* bih  = (const float*)d_in[8];
  const float* bhh  = (const float*)d_in[9];
  const float* fcw  = (const float*)d_in[10];
  const float* fcb  = (const float*)d_in[11];
  float* out = (float*)d_out;

  constexpr size_t SMEM_HALVES =
      (size_t)MLPP * HDIM + (size_t)HDIM * MLPP + (size_t)3 * HDIM * DIN +
      (size_t)3 * HDIM * HDIM + (size_t)NCLS * HDIM +
      (size_t)WAVES_PER_BLOCK * ROWS_PER_WAVE * HDIM +
      (size_t)WAVES_PER_BLOCK * ROWS_PER_WAVE * MLPP;
  constexpr size_t SMEM_BYTES =
      SMEM_HALVES * 2 + (size_t)(MLPP + HDIM + 6 * HDIM + NCLS) * 4;

  // Idempotent, called every time (no static guards per harness contract);
  // host-side attribute set, not a stream op, so graph-capture safe.
  hipFuncSetAttribute((const void*)ode_rnn_kernel,
                      hipFuncAttributeMaxDynamicSharedMemorySize,
                      (int)SMEM_BYTES);

  dim3 grid(BATCH / ROWS_PER_BLOCK);   // 32 workgroups x 4 waves x 16 rows
  dim3 block(128);
  ode_rnn_kernel<<<grid, block, SMEM_BYTES, stream>>>(
      x, tarr, w1, b1, w2, b2, wih, whh, bih, bhh, fcw, fcb, out);
}
